// LSTM_78451872628899
// MI455X (gfx1250) — matrix operand order repaired and checked by reading, not compiled, not run on hardware
//
#include <hip/hip_runtime.h>

// ---------------------------------------------------------------------------
// LSTM for MI455X (gfx1250, wave32, WMMA).
// B=64 T=1024 D=H=O=512, C=1024, 4H=2048.
//
// Serial critical path = h(64x512) @ W_h(512x2048) per step. 16 persistent
// workgroups = 4 batch tiles (M=16) x 4 gate-column groups (512 z-cols each,
// gates interleaved per j-block so the c/h update is WG-local). Per-step
// cross-WG h exchange via double-buffered global buffer + release/acquire
// counter. All GEMMs use v_wmma_f32_16x16x32_bf16 with f32 accumulation.
// x-part of the gate GEMM (Zx) and the output projection are big parallel
// bf16 WMMA GEMMs. Weight leading dims are compile-time so B-frag loads are
// base+immediate; B bases are hoisted (time-invariant in the recurrence);
// Zx rows are prefetched at step start to overlap HBM latency with the GEMM.
// ---------------------------------------------------------------------------

typedef __bf16 bf16;
typedef __bf16 bf16x16 __attribute__((ext_vector_type(16)));
typedef __bf16 bf16x8  __attribute__((ext_vector_type(8)));
typedef float  f32x8   __attribute__((ext_vector_type(8)));

union Frag16 { bf16x16 v; bf16x8 h[2]; };

#define NB   64
#define NT   1024
#define NK   512      // K for every GEMM (h-part, x-part, projection)
#define N4H  2048

__device__ __forceinline__ float sigmf(float x) { return 1.0f / (1.0f + __expf(-x)); }
__device__ __forceinline__ float tanhfast(float x) { return 2.0f / (1.0f + __expf(-2.0f * x)) - 1.0f; }

// -------------------------------------------------------------------- prep --
__global__ __launch_bounds__(256) void prep_init(const float* __restrict__ hinit,
                                                 int* __restrict__ flags,
                                                 bf16* __restrict__ hbuf) {
  int i = blockIdx.x * 256 + threadIdx.x;          // over 64*512 = 32768
  if (i < NB * NK) {
    hbuf[i]            = (bf16)hinit[i];           // parity-0 buffer = h_init
    hbuf[NB * NK + i]  = (bf16)0.0f;               // parity-1 buffer
  }
  if (i < 4) flags[i] = 0;
}

// Pack W_all transposed & gate-interleaved:  Wt[n'][k], n' = jgrp*512 + gate*128 + (j&127)
__global__ __launch_bounds__(256) void prep_pack(const float* __restrict__ Wf,
                                                 const float* __restrict__ Wi,
                                                 const float* __restrict__ Wc,
                                                 const float* __restrict__ Wo,
                                                 const float* __restrict__ bfv,
                                                 const float* __restrict__ biv,
                                                 const float* __restrict__ bcv,
                                                 const float* __restrict__ bov,
                                                 const float* __restrict__ Why,
                                                 bf16* __restrict__ Wt,
                                                 float* __restrict__ bpack,
                                                 bf16* __restrict__ Whyt) {
  size_t idx = (size_t)blockIdx.x * 256 + threadIdx.x;   // over 2048*1024
  if (idx < (size_t)N4H * 1024) {
    int n = (int)(idx >> 10), k = (int)(idx & 1023);
    int cgrp = n >> 9, r = n & 511, gate = r >> 7, j = (cgrp << 7) + (r & 127);
    const float* W = (gate == 0) ? Wf : (gate == 1) ? Wi : (gate == 2) ? Wc : Wo;
    Wt[idx] = (bf16)W[(size_t)k * 512 + j];              // W_* is (C=1024, H=512) row-major
  }
  if (idx < N4H) {
    int n = (int)idx;
    int cgrp = n >> 9, r = n & 511, gate = r >> 7, j = (cgrp << 7) + (r & 127);
    const float* bb = (gate == 0) ? bfv : (gate == 1) ? biv : (gate == 2) ? bcv : bov;
    bpack[idx] = bb[j];
  }
  if (idx < (size_t)512 * 512) {                          // Whyt[o][h] = Why[h][o]
    int o = (int)(idx >> 9), h = (int)(idx & 511);
    Whyt[idx] = (bf16)Why[(size_t)h * 512 + o];
  }
}

__global__ __launch_bounds__(256) void prep_cvt_x(const float* __restrict__ x,
                                                  bf16* __restrict__ xbf) {
  size_t i = (size_t)blockIdx.x * 256 + threadIdx.x;      // over 64*1024*512
  xbf[i] = (bf16)x[i];
}

// -------------------------------------------------- generic bf16 WMMA GEMM --
// Out[m, n] = sum_k A[m, k] * Wt[n, KOFF + k] + bias[n]
// A: M x 512 bf16 row-major.  Wt: N-major, WLD elements per row (compile-time).
// Block: 256 threads = 8 waves; 32 rows x 512 cols per block (grid.y = col blocks).
template <int WLD, int KOFF>
__global__ __launch_bounds__(256) void gemm_bf16(const bf16* __restrict__ A,
                                                 const bf16* __restrict__ Wt,
                                                 const float* __restrict__ bias,
                                                 float* __restrict__ Out,
                                                 int ldOut) {
  __shared__ bf16 As[32 * NK];
  const int tid   = threadIdx.x;
  const int lane  = tid & 31;
  const int wave  = tid >> 5;
  const int mlane = lane & 15;
  const int hi    = lane >> 4;
  const int m0    = blockIdx.x * 32;
  const int ngrp  = blockIdx.y * 512;

  // stage 32x512 bf16 A tile (32 KB) into LDS
  const bf16* Ablk = A + (size_t)m0 * NK;
  for (int i = tid; i < 32 * 64; i += 256)
    *(bf16x8*)(As + i * 8) = *(const bf16x8*)(Ablk + (size_t)i * 8);
  __syncthreads();

  // B-fragment base pointers (constant over the k loop): base + ks*32 elems.
  const bf16* bbase[4];
#pragma unroll
  for (int nt = 0; nt < 4; ++nt)
    bbase[nt] = Wt + (size_t)(ngrp + wave * 64 + nt * 16 + mlane) * WLD + KOFF + hi * 16;

  f32x8 acc[2][4] = {};
#pragma unroll 4
  for (int ks = 0; ks < 16; ++ks) {
    Frag16 a0, a1;
    int ko = ks * 32 + hi * 16;
    a0.h[0] = *(const bf16x8*)&As[(mlane)      * NK + ko];
    a0.h[1] = *(const bf16x8*)&As[(mlane)      * NK + ko + 8];
    a1.h[0] = *(const bf16x8*)&As[(16 + mlane) * NK + ko];
    a1.h[1] = *(const bf16x8*)&As[(16 + mlane) * NK + ko + 8];
#pragma unroll
    for (int nt = 0; nt < 4; ++nt) {
      bf16x16 b = *(const bf16x16*)(bbase[nt] + ks * 32);
      acc[0][nt] = __builtin_amdgcn_wmma_f32_16x16x32_bf16(false, a0.v, false, b,
                                                           (short)0, acc[0][nt], false, false);
      acc[1][nt] = __builtin_amdgcn_wmma_f32_16x16x32_bf16(false, a1.v, false, b,
                                                           (short)0, acc[1][nt], false, false);
    }
  }
#pragma unroll
  for (int ms = 0; ms < 2; ++ms)
#pragma unroll
    for (int nt = 0; nt < 4; ++nt) {
      int col = ngrp + wave * 64 + nt * 16 + mlane;
      float bv = bias[col];
      float* op = Out + (size_t)(m0 + ms * 16 + hi * 8) * ldOut + col;
#pragma unroll
      for (int r = 0; r < 8; ++r)
        op[(size_t)r * ldOut] = acc[ms][nt][r] + bv;
    }
}

// --------------------------------------------------------- recurrence core --
// 16 persistent WGs: blockIdx.x = g + 4*cg, g = batch tile (16 rows), cg = j-group.
__global__ __launch_bounds__(256) void lstm_rec(const bf16* __restrict__ Wt,     // 2048 x 1024
                                                const float* __restrict__ Zx,    // (B*T) x 2048
                                                const float* __restrict__ cinit, // 64 x 512
                                                bf16* __restrict__ hbuf,         // [2][64][512]
                                                bf16* __restrict__ hs,           // (B*T) x 512
                                                float* __restrict__ hfin,        // 64 x 512
                                                float* __restrict__ cfin,        // 64 x 512
                                                int* __restrict__ flags) {       // [4]
  __shared__ bf16  Hs[16 * NK];    // staged h_prev tile (16 KB)
  __shared__ float Zs[16 * 512];   // z slice exchange (32 KB)

  const int tid   = threadIdx.x;
  const int lane  = tid & 31;
  const int wave  = tid >> 5;
  const int mlane = lane & 15;
  const int hi    = lane >> 4;
  const int g     = blockIdx.x & 3;
  const int cg    = blockIdx.x >> 2;

  // state ownership: j_local = tid&127, batch rows rb0..rb0+7
  const int jl  = tid & 127;
  const int rb0 = (tid >> 7) * 8;
  const int jg  = cg * 128 + jl;

  float cst[8];
#pragma unroll
  for (int r = 0; r < 8; ++r)
    cst[r] = cinit[(size_t)(g * 16 + rb0 + r) * 512 + jg];

  // Time-invariant B-fragment base pointers (weights stream from L2 each step).
  const bf16* bbase[4];
#pragma unroll
  for (int nt = 0; nt < 4; ++nt)
    bbase[nt] = Wt + (size_t)(cg * 512 + wave * 64 + nt * 16 + mlane) * 1024 + hi * 16;

  // Incremental per-thread row pointers for Zx reads and h stores.
  const float* zxp[8];
  bf16*        hsp[8];
#pragma unroll
  for (int r = 0; r < 8; ++r) {
    int b  = g * 16 + rb0 + r;
    zxp[r] = Zx + (size_t)b * NT * N4H + (size_t)cg * 512 + jl;
    hsp[r] = hs + (size_t)b * NT * NK + jg;
  }
  bf16* hw0 = hbuf + (size_t)g * 16 * NK + (size_t)rb0 * NK + jg;            // parity 0
  bf16* hw1 = hw0 + (size_t)NB * NK;                                         // parity 1
  const bf16* hr0 = hbuf + (size_t)g * 16 * NK;

  for (int t = 0; t < NT; ++t) {
    // --- stage h_{t} (parity t&1) ---
    const bf16* hr = hr0 + (size_t)(t & 1) * NB * NK;
    for (int i = tid; i < 16 * 64; i += 256)
      *(bf16x8*)(Hs + i * 8) = *(const bf16x8*)(hr + (size_t)i * 8);

    // Prefetch this step's Zx rows (consumed after the GEMM) so the HBM fetch
    // overlaps the WMMA work. One lane per 128B line.
    if ((jl & 31) == 0) {
#pragma unroll
      for (int r = 0; r < 8; ++r) {
        __builtin_prefetch(zxp[r] +   0, 0, 1);
        __builtin_prefetch(zxp[r] + 128, 0, 1);
        __builtin_prefetch(zxp[r] + 256, 0, 1);
        __builtin_prefetch(zxp[r] + 384, 0, 1);
      }
    }
    __syncthreads();

    // --- z slice = h_tile(16x512) @ Wh[cg] (512x512) ---
    f32x8 acc[4] = {};
#pragma unroll 4
    for (int ks = 0; ks < 16; ++ks) {
      Frag16 a;
      int ko = ks * 32 + hi * 16;
      a.h[0] = *(const bf16x8*)&Hs[mlane * NK + ko];
      a.h[1] = *(const bf16x8*)&Hs[mlane * NK + ko + 8];
#pragma unroll
      for (int nt = 0; nt < 4; ++nt) {
        bf16x16 b = *(const bf16x16*)(bbase[nt] + ks * 32);
        acc[nt] = __builtin_amdgcn_wmma_f32_16x16x32_bf16(false, a.v, false, b,
                                                          (short)0, acc[nt], false, false);
      }
    }
#pragma unroll
    for (int nt = 0; nt < 4; ++nt) {
      int col = wave * 64 + nt * 16 + mlane;
#pragma unroll
      for (int r = 0; r < 8; ++r)
        Zs[(hi * 8 + r) * 512 + col] = acc[nt][r];
    }
    __syncthreads();

    // --- gates + state update for owned (b, j) cells ---
    bf16* hw = (t & 1) ? hw0 : hw1;   // write parity (t+1)&1
#pragma unroll
    for (int r = 0; r < 8; ++r) {
      int m = rb0 + r;
      float zf = Zs[m * 512 +   0 + jl] + zxp[r][  0];
      float zi = Zs[m * 512 + 128 + jl] + zxp[r][128];
      float zg = Zs[m * 512 + 256 + jl] + zxp[r][256];
      float zo = Zs[m * 512 + 384 + jl] + zxp[r][384];
      float fg = sigmf(zf), ig = sigmf(zi), gg = tanhfast(zg), og = sigmf(zo);
      float c  = fg * cst[r] + ig * gg;
      cst[r]   = c;
      float h  = og * tanhfast(c);
      hw[(size_t)r * NK] = (bf16)h;
      *hsp[r] = (bf16)h;
      if (t == NT - 1) {
        int b = g * 16 + m;
        hfin[(size_t)b * 512 + jg] = h;
        cfin[(size_t)b * 512 + jg] = c;
      }
      zxp[r] += N4H;
      hsp[r] += NK;
    }
    __threadfence();    // make h visible at agent scope
    __syncthreads();

    // --- cross-WG step barrier among the 4 j-groups of this batch tile ---
    if (tid == 0) {
      __hip_atomic_fetch_add(&flags[g], 1, __ATOMIC_RELEASE, __HIP_MEMORY_SCOPE_AGENT);
      int target = 4 * (t + 1);
      while (__hip_atomic_load(&flags[g], __ATOMIC_ACQUIRE, __HIP_MEMORY_SCOPE_AGENT) < target)
        __builtin_amdgcn_s_sleep(1);
    }
    __syncthreads();
  }
}

// ------------------------------------------------------------------ launch --
extern "C" void kernel_launch(void* const* d_in, const int* in_sizes, int n_in,
                              void* d_out, int out_size, void* d_ws, size_t ws_size,
                              hipStream_t stream) {
  (void)in_sizes; (void)n_in; (void)out_size; (void)ws_size;

  const float* x     = (const float*)d_in[0];
  const float* hinit = (const float*)d_in[1];
  const float* cinit = (const float*)d_in[2];
  const float* Wf    = (const float*)d_in[3];
  const float* bfv   = (const float*)d_in[4];
  const float* Wi    = (const float*)d_in[5];
  const float* biv   = (const float*)d_in[6];
  const float* Wc    = (const float*)d_in[7];
  const float* bcv   = (const float*)d_in[8];
  const float* Wo    = (const float*)d_in[9];
  const float* bov   = (const float*)d_in[10];
  const float* Why   = (const float*)d_in[11];
  const float* bhy   = (const float*)d_in[12];

  float* out  = (float*)d_out;                       // (B,T,O) f32
  float* hfin = out + (size_t)NB * NT * 512;
  float* cfin = hfin + (size_t)NB * 512;

  // workspace carve-up (256 B aligned)
  char* ws = (char*)d_ws;
  size_t off = 0;
  auto take = [&](size_t bytes) { char* p = ws + off; off += (bytes + 255) & ~(size_t)255; return p; };
  bf16*  Wt    = (bf16*)take((size_t)N4H * 1024 * sizeof(bf16));   // 4 MB
  bf16*  Whyt  = (bf16*)take((size_t)512 * 512 * sizeof(bf16));    // 0.5 MB
  float* bpack = (float*)take((size_t)N4H * sizeof(float));
  bf16*  xbf   = (bf16*)take((size_t)NB * NT * 512 * sizeof(bf16));    // 64 MB
  bf16*  hsbuf = (bf16*)take((size_t)NB * NT * 512 * sizeof(bf16));    // 64 MB
  float* Zx    = (float*)take((size_t)NB * NT * N4H * sizeof(float));  // 512 MB
  bf16*  hbuf  = (bf16*)take((size_t)2 * NB * NK * sizeof(bf16));
  int*   flags = (int*)take(256);

  // 1. init h buffers + step counters
  prep_init<<<dim3((NB * NK + 255) / 256), dim3(256), 0, stream>>>(hinit, flags, hbuf);

  // 2. pack weights (transpose to N-major bf16, gate-interleaved) + biases
  prep_pack<<<dim3((N4H * 1024) / 256), dim3(256), 0, stream>>>(
      Wf, Wi, Wc, Wo, bfv, biv, bcv, bov, Why, Wt, bpack, Whyt);

  // 3. x -> bf16
  prep_cvt_x<<<dim3((NB * NT * 512) / 256), dim3(256), 0, stream>>>(x, xbf);

  // 4. Zx = x @ W_x + b  (parallel GEMM, M=65536, N=2048, K=512; KOFF=512)
  gemm_bf16<1024, 512><<<dim3((NB * NT) / 32, 4), dim3(256), 0, stream>>>(
      xbf, Wt, bpack, Zx, /*ldOut=*/N4H);

  // 5. serial recurrence: 16 persistent WGs
  lstm_rec<<<dim3(16), dim3(256), 0, stream>>>(Wt, Zx, cinit, hbuf, hsbuf, hfin, cfin, flags);

  // 6. outputs = hs @ W_hy + b_hy  (parallel GEMM, M=65536, N=512, K=512)
  gemm_bf16<512, 0><<<dim3((NB * NT) / 32, 1), dim3(256), 0, stream>>>(
      hsbuf, Whyt, bhy, out, /*ldOut=*/512);
}
